// DynamicModel_31370441130299
// MI455X (gfx1250) — matrix operand order repaired and checked by reading, not compiled, not run on hardware
//
#include <hip/hip_runtime.h>
#include <hip/hip_bf16.h>

// ---------------------------------------------------------------------------
// Tsit5 neural-ODE step on MI455X (gfx1250), wave32 + v_wmma_f32_16x16x32_f16.
//
// prep_weights_kernel : f32 weights -> f16 in WMMA-B fragment order (d_ws)
// tsit5_mlp_kernel    : 1024 blocks x 256 thr; each wave owns 16 batch rows.
//   State kept in WMMA C-layout registers with forward accumulation.
//   GEMMs use a rotating modulo software pipeline (register-reuse WAR deps
//   pin B-fragment loads DEPTH-1 WMMAs ahead) and a rolling row pointer so
//   every ds_load offset is a small immediate (one live address register).
// ---------------------------------------------------------------------------

typedef __attribute__((ext_vector_type(16))) _Float16 v16h;
typedef __attribute__((ext_vector_type(8)))  float    v8f;

#define OBS_DIM 64
#define ACT_DIM 16
#define HID     256
#define MROWS   128          // batch rows per block (8 waves x 16 rows)
#define XK      96           // GEMM1 K padded: 64 obs + 16 act + 16 zeros
#define DT      0.05

// Fragment = 512 halfs (32x16 f16 B tile), layout [lane][16 halfs]
#define W1_FRAGS (3 * 16)    // ksteps x ntiles
#define W2_FRAGS (8 * 16)
#define W3_FRAGS (8 * 4)
#define W_HALFS  ((W1_FRAGS + W2_FRAGS + W3_FRAGS) * 512)   // 106496
#define W_BYTES  (W_HALFS * 2)                               // 212992
#define W1_OFF   0
#define W2_OFF   (W1_FRAGS * 512)                            // 24576
#define W3_OFF   ((W1_FRAGS + W2_FRAGS) * 512)               // 90112

#define BIAS_FLOATS 576                                      // 256+256+64
#define BIAS_BYTES  (BIAS_FLOATS * 4)                        // 2304
#define WS_BYTES    (W_BYTES + BIAS_BYTES)                   // 215296

// LDS layout (bytes)
#define L_BIAS_OFF  W_BYTES                                  // 212992
#define L_X_OFF     (L_BIAS_OFF + BIAS_BYTES)                // 215296
#define L_X_BYTES   (MROWS * XK * 2)                         // 24576
#define L_H_OFF     (L_X_OFF + L_X_BYTES)                    // 239872
#define L_H_BYTES   (MROWS * HID * 2)                        // 65536
#define LDS_TOTAL   (L_H_OFF + L_H_BYTES)                    // 305408 <= 320KB

// Tsit5 tableau
__device__ constexpr double TS_A[6][5] = {
    {0, 0, 0, 0, 0},
    {0.161, 0, 0, 0, 0},
    {-0.008480655492356989, 0.335480655492357, 0, 0, 0},
    {2.8971530571054935, -6.359448489975075, 4.3622954328695815, 0, 0},
    {5.325864828439257, -11.748883564062828, 7.4955393428898365,
     -0.09249506636175525, 0},
    {5.86145544294642, -12.92096931784711, 8.159367898576159,
     -0.071584973281401, -0.028269050394068383},
};
__device__ constexpr double TS_B[6] = {
    0.09646076681806523, 0.01, 0.4798896504144996,
    1.379008574103742, -3.290069515436081, 2.324710524099774,
};

// ---------------------------------------------------------------------------
// Prep: W (f32, KxN row-major) -> f16 fragments.
// Fragment element (lane, h): K = kk*32 + (lane>=16)*16 + h, N = nt*16 + lane%16
// ---------------------------------------------------------------------------
__global__ __launch_bounds__(256) void prep_weights_kernel(
    const float* __restrict__ W1, const float* __restrict__ b1,
    const float* __restrict__ W2, const float* __restrict__ b2,
    const float* __restrict__ W3, const float* __restrict__ b3,
    _Float16* __restrict__ wh, float* __restrict__ bf)
{
    int gid = blockIdx.x * 256 + threadIdx.x;
    if (gid < W_HALFS) {
        int i = gid;
        const float* W;
        int ntiles, ksrc, nsrc;
        if (i < W2_OFF)        { W = W1; ntiles = 16; ksrc = 80;  nsrc = 256; }
        else if (i < W3_OFF)   { i -= W2_OFF; W = W2; ntiles = 16; ksrc = 256; nsrc = 256; }
        else                   { i -= W3_OFF; W = W3; ntiles = 4;  ksrc = 256; nsrc = 64;  }
        int f    = i >> 9;
        int r    = i & 511;
        int lane = r >> 4;
        int h    = r & 15;
        int kk   = f / ntiles;
        int nt   = f % ntiles;
        int k    = kk * 32 + ((lane >> 4) << 4) + h;
        int n    = nt * 16 + (lane & 15);
        float v  = (k < ksrc) ? W[k * nsrc + n] : 0.0f;
        wh[gid]  = (_Float16)v;
    }
    if (gid < BIAS_FLOATS) {
        float v = (gid < 256) ? b1[gid]
                : (gid < 512) ? b2[gid - 256]
                              : b3[gid - 512];
        bf[gid] = v;
    }
}

// ---------------------------------------------------------------------------
// Fragment loaders (two aligned ds_load_b128 each).
// ---------------------------------------------------------------------------
union F16x16 { uint4 u[2]; v16h h; };

__device__ __forceinline__ F16x16 load_afrag(const _Float16* base, int astride,
                                             int r, int sel, int kk)
{
    const _Float16* p = base + r * astride + kk * 32 + sel;
    F16x16 f;
    f.u[0] = *(const uint4*)(p);
    f.u[1] = *(const uint4*)(p + 8);
    return f;
}

// load B tile nt relative to a rolling row pointer (offsets <= 15KB)
__device__ __forceinline__ F16x16 load_brow(const _Float16* rowlane, int nt)
{
    const _Float16* p = rowlane + nt * 512;
    F16x16 f;
    f.u[0] = *(const uint4*)(p);
    f.u[1] = *(const uint4*)(p + 8);
    return f;
}

// ---------------------------------------------------------------------------
// Rotating-modulo software-pipelined GEMM:
// 16 rows x (KSTEPS*32) @ fragments -> NT acc tiles.
// Step i consumes B slot i%DEPTH, then reloads that slot with fragment
// i+DEPTH (WAR dep pins the load ahead of its consumer). The row pointer
// advances lazily when the prefetch crosses a k-row, keeping all ds offsets
// as small immediates off one live register.
// ---------------------------------------------------------------------------
template <int KSTEPS, int NT, int FSTRIDE, int DEPTH>
__device__ __forceinline__ void gemm_tiles(const _Float16* __restrict__ Arows,
                                           int astride,
                                           const _Float16* __restrict__ Wfrag,
                                           int lane, v8f* acc)
{
    const int r   = lane & 15;
    const int sel = (lane >> 4) << 4;   // 0 or 16
    constexpr int TOTAL    = KSTEPS * NT;
    constexpr int ROWHALFS = FSTRIDE * 512;
    static_assert(DEPTH <= NT, "prologue must stay within row 0");

    const _Float16* wpf = Wfrag + lane * 16;   // rolling row base (lane folded)
    F16x16 bs[DEPTH];
#pragma unroll
    for (int i = 0; i < DEPTH; ++i) bs[i] = load_brow(wpf, i);

    F16x16 a  = load_afrag(Arows, astride, r, sel, 0);
    F16x16 an = a;
    if (KSTEPS > 1) an = load_afrag(Arows, astride, r, sel, 1);

#pragma unroll
    for (int i = 0; i < TOTAL; ++i) {
        const int nt = i % NT;
        acc[nt] = __builtin_amdgcn_wmma_f32_16x16x32_f16(
            false, a.h, false, bs[i % DEPTH].h, (short)0, acc[nt],
            false, false);
        const int pf = i + DEPTH;
        if (pf < TOTAL) {
            if (pf % NT == 0) wpf += ROWHALFS;   // entered next k-row
            bs[i % DEPTH] = load_brow(wpf, pf % NT);
        }
        if (nt == NT - 1) {
            a = an;
            if (i / NT + 2 < KSTEPS)
                an = load_afrag(Arows, astride, r, sel, i / NT + 2);
        }
    }
}

// silu with HW reciprocal (v_rcp_f32): 5 VALU, short live ranges
__device__ __forceinline__ float silu(float x)
{
    return x * __builtin_amdgcn_rcpf(1.0f + __expf(-x));
}

// ---------------------------------------------------------------------------
// One Tsit5 stage, fully compile-time specialized.
// Stage 0 *creates* ys[1..5]/y1 from y0 + c*k0; later stages accumulate.
// ---------------------------------------------------------------------------
template <int S>
__device__ __forceinline__ void run_stage(
    v8f (&ys)[6][4], v8f (&y1v)[4],
    const _Float16* __restrict__ Wl, const float* __restrict__ Bl,
    _Float16* __restrict__ Xw, _Float16* __restrict__ Hw,
    int lane, int mrow, int ncol)
{
    const v8f vzero = {};

    // stage input -> X[:,0:64] as f16 (C layout -> row major)
#pragma unroll
    for (int t = 0; t < 4; ++t)
#pragma unroll
        for (int r = 0; r < 8; ++r)
            Xw[(mrow + r) * XK + t * 16 + ncol] = (_Float16)ys[S][t][r];

    // ---- GEMM1: X[16x96] @ W1 -> H1[16x256], silu (two 8-tile halves) ------
#pragma unroll
    for (int hh = 0; hh < 2; ++hh) {
        v8f acc[8];
#pragma unroll
        for (int i = 0; i < 8; ++i) acc[i] = vzero;
        gemm_tiles<3, 8, 16, 6>(Xw, XK, Wl + W1_OFF + hh * 8 * 512, lane, acc);
#pragma unroll
        for (int nt = 0; nt < 8; ++nt) {
            const int col = (hh * 8 + nt) * 16 + ncol;
            const float bb = Bl[col];
#pragma unroll
            for (int r = 0; r < 8; ++r)
                Hw[(mrow + r) * HID + col] = (_Float16)silu(acc[nt][r] + bb);
        }
    }

    // ---- GEMM2: H1[16x256] @ W2 -> H2[16x256], silu ------------------------
    {
        v8f acc[16];
#pragma unroll
        for (int i = 0; i < 16; ++i) acc[i] = vzero;
        gemm_tiles<8, 16, 16, 6>(Hw, HID, Wl + W2_OFF, lane, acc);
        // all H1 reads complete before overwriting (in-order per-wave LDS)
#pragma unroll
        for (int nt = 0; nt < 16; ++nt) {
            const int col = nt * 16 + ncol;
            const float bb = Bl[256 + col];
#pragma unroll
            for (int r = 0; r < 8; ++r)
                Hw[(mrow + r) * HID + col] = (_Float16)silu(acc[nt][r] + bb);
        }
    }

    // ---- GEMM3: H2[16x256] @ W3 -> k[16x64] --------------------------------
    v8f kv[4];
#pragma unroll
    for (int i = 0; i < 4; ++i) kv[i] = vzero;
    gemm_tiles<8, 4, 4, 4>(Hw, HID, Wl + W3_OFF, lane, kv);
#pragma unroll
    for (int t = 0; t < 4; ++t) {
        const float bb = Bl[512 + t * 16 + ncol];
#pragma unroll
        for (int r = 0; r < 8; ++r) kv[t][r] += bb;
    }

    // ---- forward-accumulate k_S into future stage inputs and y1 ------------
#pragma unroll
    for (int i = S + 1; i < 6; ++i) {
        const float c = (float)(DT * TS_A[i][S]);
#pragma unroll
        for (int t = 0; t < 4; ++t)
#pragma unroll
            for (int r = 0; r < 8; ++r) {
                if (S == 0) ys[i][t][r] = ys[0][t][r] + c * kv[t][r];
                else        ys[i][t][r] += c * kv[t][r];
            }
    }
    const float cb = (float)(DT * TS_B[S]);
#pragma unroll
    for (int t = 0; t < 4; ++t)
#pragma unroll
        for (int r = 0; r < 8; ++r) {
            if (S == 0) y1v[t][r] = ys[0][t][r] + cb * kv[t][r];
            else        y1v[t][r] += cb * kv[t][r];
        }
}

// ---------------------------------------------------------------------------
// Main kernel.
// ---------------------------------------------------------------------------
__global__ __launch_bounds__(256, 1) void tsit5_mlp_kernel(
    const float* __restrict__ obs, const float* __restrict__ act,
    const uint4* __restrict__ wsrc, float* __restrict__ out)
{
    extern __shared__ __align__(16) char smem[];
    _Float16* Wl = (_Float16*)smem;
    float*    Bl = (float*)(smem + L_BIAS_OFF);
    _Float16* Xl = (_Float16*)(smem + L_X_OFF);
    _Float16* Hl = (_Float16*)(smem + L_H_OFF);

    const int tid  = threadIdx.x;
    const int lane = tid & 31;
    const int wv   = tid >> 5;
    const int row0 = blockIdx.x * MROWS;

    // WMMA C-layout coords
    const int mrow = (lane >> 4) << 3;   // 0 or 8
    const int ncol = lane & 15;
    const int wr0  = wv * 16;
    _Float16* Xw = Xl + wr0 * XK;
    _Float16* Hw = Hl + wr0 * HID;

    // --- y0 loads first: HBM latency overlaps the LDS weight staging --------
    v8f ys[6][4], y1v[4];
#pragma unroll
    for (int t = 0; t < 4; ++t)
#pragma unroll
        for (int r = 0; r < 8; ++r)
            ys[0][t][r] =
                obs[(size_t)(row0 + wr0 + mrow + r) * OBS_DIM + t * 16 + ncol];

    // --- stage weights + biases into LDS (coalesced b128) -------------------
    {
        uint4* dst = (uint4*)smem;
        const int n16 = WS_BYTES / 16;          // 13456
        for (int i = tid; i < n16; i += 256) dst[i] = wsrc[i];
    }
    // --- actions -> X[:,64:80] f16; zero-pad X[:,80:96] ----------------------
    if (tid < MROWS) {
        const float* ap = act + (size_t)(row0 + tid) * ACT_DIM;
        _Float16* xp = Xl + tid * XK + OBS_DIM;
#pragma unroll
        for (int j = 0; j < ACT_DIM; ++j) xp[j] = (_Float16)ap[j];
#pragma unroll
        for (int j = 0; j < 16; ++j) xp[ACT_DIM + j] = (_Float16)0.0f;
    }
    __syncthreads();   // only barrier: afterwards each wave touches own rows

    run_stage<0>(ys, y1v, Wl, Bl, Xw, Hw, lane, mrow, ncol);
    run_stage<1>(ys, y1v, Wl, Bl, Xw, Hw, lane, mrow, ncol);
    run_stage<2>(ys, y1v, Wl, Bl, Xw, Hw, lane, mrow, ncol);
    run_stage<3>(ys, y1v, Wl, Bl, Xw, Hw, lane, mrow, ncol);
    run_stage<4>(ys, y1v, Wl, Bl, Xw, Hw, lane, mrow, ncol);
    run_stage<5>(ys, y1v, Wl, Bl, Xw, Hw, lane, mrow, ncol);

    // --- write y1 ------------------------------------------------------------
#pragma unroll
    for (int t = 0; t < 4; ++t)
#pragma unroll
        for (int r = 0; r < 8; ++r)
            out[(size_t)(row0 + wr0 + mrow + r) * OBS_DIM + t * 16 + ncol] =
                y1v[t][r];
}

// ---------------------------------------------------------------------------
extern "C" void kernel_launch(void* const* d_in, const int* in_sizes, int n_in,
                              void* d_out, int out_size, void* d_ws,
                              size_t ws_size, hipStream_t stream)
{
    (void)in_sizes; (void)n_in; (void)out_size; (void)ws_size;
    const float* obs = (const float*)d_in[0];
    const float* act = (const float*)d_in[1];
    const float* W1  = (const float*)d_in[2];
    const float* b1  = (const float*)d_in[3];
    const float* W2  = (const float*)d_in[4];
    const float* b2  = (const float*)d_in[5];
    const float* W3  = (const float*)d_in[6];
    const float* b3  = (const float*)d_in[7];

    _Float16* wh = (_Float16*)d_ws;
    float*    bf = (float*)((char*)d_ws + W_BYTES);

    (void)hipFuncSetAttribute((const void*)tsit5_mlp_kernel,
                              hipFuncAttributeMaxDynamicSharedMemorySize,
                              LDS_TOTAL);

    prep_weights_kernel<<<(W_HALFS + 255) / 256, 256, 0, stream>>>(
        W1, b1, W2, b2, W3, b3, wh, bf);

    const int nblocks = 131072 / MROWS;   // 1024
    tsit5_mlp_kernel<<<nblocks, 256, LDS_TOTAL, stream>>>(
        obs, act, (const uint4*)d_ws, (float*)d_out);
}
